// NeuralKYAttention_31138512896504
// MI455X (gfx1250) — matrix operand order repaired and checked by reading, not compiled, not run on hardware
//
#include <hip/hip_runtime.h>
#include <hip/hip_bf16.h>
#include <cmath>
#include <cstdint>

#define BB 2
#define LL 2048
#define DD 1024
#define HH 16
#define WW 32
#define HD 64

typedef __attribute__((ext_vector_type(16))) __bf16 v16bf;
typedef __attribute__((ext_vector_type(8)))  __bf16 v8bf;
typedef __attribute__((ext_vector_type(8)))  float  v8f;
typedef unsigned int u32x4 __attribute__((ext_vector_type(4)));
typedef int          i32x8 __attribute__((ext_vector_type(8)));
typedef int          i32x4 __attribute__((ext_vector_type(4)));

__device__ __forceinline__ v8f zero8f() {
    v8f r = {0.f,0.f,0.f,0.f,0.f,0.f,0.f,0.f};
    return r;
}
__device__ __forceinline__ v8bf zero8b() {
    v8bf r;
#pragma unroll
    for (int i = 0; i < 8; ++i) r[i] = (__bf16)0.f;
    return r;
}
__device__ __forceinline__ v16bf cat16(v8bf lo, v8bf hi) {
    v16bf r;
#pragma unroll
    for (int i = 0; i < 8; ++i) { r[i] = lo[i]; r[8 + i] = hi[i]; }
    return r;
}

// A-matrix fragment (16x32 bf16): lane holds row m=lane%16.
// Contiguous K layout (matches B): lanes 0-15 hold K kb..kb+15; lanes 16-31 hold
// K kb+16..kb+31. Memory is row-major with given row stride (elements).
__device__ __forceinline__ v16bf load_a_frag(const __bf16* base, int rowStride,
                                             int kb, int lane) {
    int m  = lane & 15;
    int hi = lane >> 4;
    const __bf16* r = base + (size_t)m * rowStride + kb;
    v8bf lo8 = *(const v8bf*)(r + (hi ? 16 : 0));
    v8bf hi8 = *(const v8bf*)(r + (hi ? 24 : 8));
    return cat16(lo8, hi8);
}
// B-matrix fragment (32x16 bf16) where memory holds B^T (row n has K contiguous):
// lane n=lane%16; lanes 0-15 hold K kb..kb+15, lanes 16-31 hold kb+16..kb+31.
__device__ __forceinline__ v16bf load_bt_frag(const __bf16* base, int rowStride,
                                              int kb, int lane) {
    int n  = lane & 15;
    int hi = lane >> 4;
    const __bf16* r = base + (size_t)n * rowStride + kb + (hi ? 16 : 0);
    v8bf lo8 = *(const v8bf*)(r);
    v8bf hi8 = *(const v8bf*)(r + 8);
    return cat16(lo8, hi8);
}

// ---------------------------------------------------------------------------
// Tensor Data Mover: 2D tile load (global -> LDS), D# per CDNA5 ISA.
// data_size=1 (2-byte elements). Walks tile_d1 lines of tile_d0 elements with
// line stride `stride0` (elements); LDS fill is contiguous.
__device__ __forceinline__ void tdm_load_2d(uint32_t lds_off, const void* gaddr,
                                            uint32_t tile_d0, uint32_t tile_d1,
                                            uint64_t stride0) {
    uint64_t ga = (uint64_t)(uintptr_t)gaddr;
    u32x4 g0;
    g0[0] = 1u;                                        // count=1, user mode
    g0[1] = lds_off;                                   // lds_addr (bytes)
    g0[2] = (uint32_t)(ga & 0xffffffffu);              // global_addr[31:0]
    g0[3] = (uint32_t)((ga >> 32) & 0x01ffffffu)       // global_addr[56:32]
            | (2u << 30);                              // type = 2 ("image")
    const uint32_t tensor_d0 = 0x7fffffffu;            // huge: no OOB clipping
    const uint32_t tensor_d1 = 0x7fffffffu;
    i32x8 g1;
    g1[0] = (int)(1u << 16);                           // wg_mask=0, data_size=1 (2B)
    g1[1] = (int)(tensor_d0 << 16);                    // tensor_dim0[15:0]
    g1[2] = (int)((tensor_d0 >> 16) | (tensor_d1 << 16));
    g1[3] = (int)((tensor_d1 >> 16) | (tile_d0 << 16));// tile_dim0
    g1[4] = (int)(tile_d1 & 0xffffu);                  // tile_dim1 | tile_dim2=0
    g1[5] = (int)(uint32_t)(stride0 & 0xffffffffu);    // dim0_stride[31:0]
    g1[6] = (int)(uint32_t)((stride0 >> 32) & 0xffffu);// dim0_stride[47:32]
    g1[7] = 0;                                         // dim1_stride (unused, 2D)
    i32x4 z4 = {0, 0, 0, 0};
    i32x8 z8 = {0, 0, 0, 0, 0, 0, 0, 0};
    __builtin_amdgcn_tensor_load_to_lds(g0, g1, z4, z4, z8, 0);
}

// ---------------------------------------------------------------------------
// fp32 -> bf16 convert
__global__ __launch_bounds__(256) void f2bf_kernel(const float* __restrict__ in,
                                                   __bf16* __restrict__ out, int n) {
    int i = blockIdx.x * 256 + threadIdx.x;
    if (i < n) out[i] = (__bf16)in[i];
}

// ---------------------------------------------------------------------------
// conn[h][w] = softmax_w( sum_k silu(pos*cw1[h,k]) * (pos*cw3[h,k]) * cw2[h,k] )
__global__ __launch_bounds__(512) void conn_kernel(const float* __restrict__ cw1,
                                                   const float* __restrict__ cw2,
                                                   const float* __restrict__ cw3,
                                                   float* __restrict__ conn) {
    __shared__ float Sm[HH][WW];
    int t = threadIdx.x;
    int h = t >> 5, w = t & 31;
    float pos = (float)w / 31.0f;
    float acc = 0.f;
#pragma unroll 4
    for (int k = 0; k < 128; ++k) {
        float a1 = pos * cw1[h * 128 + k];
        float a3 = pos * cw3[h * 128 + k];
        float si = a1 / (1.0f + __expf(-a1));
        acc += si * a3 * cw2[h * 128 + k];
    }
    Sm[h][w] = acc;
    __syncthreads();
    float mx = -3.0e38f;
#pragma unroll
    for (int i = 0; i < WW; ++i) mx = fmaxf(mx, Sm[h][i]);
    float z = 0.f;
#pragma unroll
    for (int i = 0; i < WW; ++i) z += __expf(Sm[h][i] - mx);
    conn[h * WW + w] = __expf(acc - mx) / z;
}

// ---------------------------------------------------------------------------
// C[m,n] = sum_k A[m,k] * Wm[n,k]   (X @ W^T, matching einsum bld,ed->ble)
// A: MxK bf16 row-major, Wm: NxK bf16 row-major. 128x128 block tile, K step 32.
// Tiles staged by the Tensor Data Mover (double-buffered, issued by wave 0,
// pipelined one K-step ahead; TENSORcnt is in-order so wait<=2 publishes the
// current pair while the next pair is in flight).
template <bool F32OUT>
__global__ __launch_bounds__(256) void gemm_xwT(const __bf16* __restrict__ A,
                                                const __bf16* __restrict__ Wm,
                                                void* __restrict__ Cout,
                                                int M, int N, int K) {
    __shared__ __align__(16) __bf16 As[2][128][32];
    __shared__ __align__(16) __bf16 Ws[2][128][32];
    int t = threadIdx.x;
    int m0 = blockIdx.y * 128;
    int n0 = blockIdx.x * 128;
    int wv = t >> 5, lane = t & 31;
    int tm = (wv & 3) * 2;   // this wave's first M 16-tile (covers 2)
    int tn = (wv >> 2) * 4;  // this wave's first N 16-tile (covers 4)

    v8f acc[2][4];
#pragma unroll
    for (int i = 0; i < 2; ++i)
#pragma unroll
        for (int j = 0; j < 4; ++j) acc[i][j] = zero8f();

    // prologue: DMA first K tile into buffer 0
    if (wv == 0) {
        tdm_load_2d((uint32_t)(uintptr_t)&As[0][0][0],
                    A + (size_t)m0 * K, 32, 128, (uint64_t)K);
        tdm_load_2d((uint32_t)(uintptr_t)&Ws[0][0][0],
                    Wm + (size_t)n0 * K, 32, 128, (uint64_t)K);
    }

    int buf = 0;
    for (int kt = 0; kt < K; kt += 32) {
        if (wv == 0) {
            if (kt + 32 < K) {
                tdm_load_2d((uint32_t)(uintptr_t)&As[buf ^ 1][0][0],
                            A + (size_t)m0 * K + kt + 32, 32, 128, (uint64_t)K);
                tdm_load_2d((uint32_t)(uintptr_t)&Ws[buf ^ 1][0][0],
                            Wm + (size_t)n0 * K + kt + 32, 32, 128, (uint64_t)K);
                __builtin_amdgcn_s_wait_tensorcnt(2);  // current pair landed
            } else {
                __builtin_amdgcn_s_wait_tensorcnt(0);  // last pair landed
            }
        }
        __syncthreads();  // publish buffer `buf` to all waves

        v16bf af[2], bfg[4];
#pragma unroll
        for (int i = 0; i < 2; ++i)
            af[i] = load_a_frag(&As[buf][(tm + i) * 16][0], 32, 0, lane);
#pragma unroll
        for (int j = 0; j < 4; ++j)
            bfg[j] = load_bt_frag(&Ws[buf][(tn + j) * 16][0], 32, 0, lane);
#pragma unroll
        for (int i = 0; i < 2; ++i)
#pragma unroll
            for (int j = 0; j < 4; ++j)
                acc[i][j] = __builtin_amdgcn_wmma_f32_16x16x32_bf16(
                    false, af[i], false, bfg[j], (short)0, acc[i][j], false, false);
        __syncthreads();  // everyone done with `buf` before TDM reuses it
        buf ^= 1;
    }

    int half = lane >> 4, nn = lane & 15;
#pragma unroll
    for (int i = 0; i < 2; ++i)
#pragma unroll
        for (int j = 0; j < 4; ++j)
#pragma unroll
            for (int v = 0; v < 8; ++v) {
                size_t r = (size_t)m0 + (tm + i) * 16 + v + 8 * half;
                size_t c = (size_t)n0 + (tn + j) * 16 + nn;
                if constexpr (F32OUT)
                    ((float*)Cout)[r * N + c] = acc[i][j][v];
                else
                    ((__bf16*)Cout)[r * N + c] = (__bf16)acc[i][j][v];
            }
}

// ---------------------------------------------------------------------------
// RoPE (interleaved pairs) + relayout [B,L,H*HD] -> [B,H,L,HD]; V relayout only.
__global__ __launch_bounds__(256) void rope_relayout(const __bf16* __restrict__ Qraw,
                                                     const __bf16* __restrict__ Kraw,
                                                     const __bf16* __restrict__ Vraw,
                                                     const float* __restrict__ cosT,
                                                     const float* __restrict__ sinT,
                                                     __bf16* __restrict__ Qr,
                                                     __bf16* __restrict__ Kr,
                                                     __bf16* __restrict__ Vr) {
    int idx = blockIdx.x * 256 + threadIdx.x;       // 2^21 threads
    int p = idx & 31;
    int h = (idx >> 5) & 15;
    int l = (idx >> 9) & 2047;
    int b = idx >> 20;
    size_t src = ((size_t)(b * LL + l)) * DD + h * HD + 2 * p;
    size_t dst = (((size_t)b * HH + h) * LL + l) * HD + 2 * p;
    float c = cosT[l * 32 + p], s = sinT[l * 32 + p];
    float q0 = (float)Qraw[src], q1 = (float)Qraw[src + 1];
    Qr[dst]     = (__bf16)(q0 * c - q1 * s);
    Qr[dst + 1] = (__bf16)(q0 * s + q1 * c);
    float k0 = (float)Kraw[src], k1 = (float)Kraw[src + 1];
    Kr[dst]     = (__bf16)(k0 * c - k1 * s);
    Kr[dst + 1] = (__bf16)(k0 * s + k1 * c);
    Vr[dst] = Vraw[src]; Vr[dst + 1] = Vraw[src + 1];
}

// ---------------------------------------------------------------------------
// Windowed attention: one wave per (b, h, 16-query tile).
// Scores over a 48-row K segment (s = l0-31 .. l0+16); each query row m uses the
// diagonal band j in [m, m+31]. Zero-padded rows (s<0) stay in the softmax with
// score 0, matching the reference's zero-pad semantics.
__global__ __launch_bounds__(32) void attn_kernel(const __bf16* __restrict__ Qr,
                                                  const __bf16* __restrict__ Kr,
                                                  const __bf16* __restrict__ Vr,
                                                  const float* __restrict__ connw,
                                                  __bf16* __restrict__ Oacc) {
    __shared__ __align__(16) __bf16 VT[64][64];   // V segment transposed: VT[d][j]
    __shared__ __align__(16) __bf16 P[16][64];    // final probs, A-matrix source
    __shared__ float connS[32];

    int lane = threadIdx.x;
    int id = blockIdx.x;
    int lt = id & 127;
    int h  = (id >> 7) & 15;
    int b  = id >> 11;
    int l0 = lt * 16;

    const size_t bh = (size_t)b * HH + h;
    const __bf16* Qg = Qr + bh * LL * HD;
    const __bf16* Kg = Kr + bh * LL * HD;
    const __bf16* Vg = Vr + bh * LL * HD;

    // zero LDS (VT pad rows j=48..63 and invalid P entries must be 0)
    {
        v8bf z = zero8b();
        v8bf* vt = (v8bf*)&VT[0][0];
        for (int i = lane; i < 512; i += 32) vt[i] = z;
        v8bf* pp = (v8bf*)&P[0][0];
        for (int i = lane; i < 128; i += 32) pp[i] = z;
    }
    connS[lane] = connw[h * 32 + lane];
    __syncthreads();

    // stage V segment transposed into LDS: j = 0..47 -> s = l0-31+j
    for (int j = lane; j < 48; j += 32) {
        int s = l0 - 31 + j;
        if (s >= 0 && s < LL) {
            const __bf16* vrow = Vg + (size_t)s * HD;
#pragma unroll
            for (int c = 0; c < 8; ++c) {
                v8bf vv = *(const v8bf*)(vrow + c * 8);
#pragma unroll
                for (int e = 0; e < 8; ++e) VT[c * 8 + e][j] = vv[e];
            }
        }
    }

    // Q A-fragments (K = 0..31 and 32..63 of head dim)
    v16bf aq0 = load_a_frag(Qg + (size_t)l0 * HD, HD, 0, lane);
    v16bf aq1 = load_a_frag(Qg + (size_t)l0 * HD, HD, 32, lane);

    // scores: S[t] = Q x Kseg^T for segment columns j = t*16 .. t*16+15
    v8f S[3];
#pragma unroll
    for (int tI = 0; tI < 3; ++tI) S[tI] = zero8f();
#pragma unroll
    for (int tI = 0; tI < 3; ++tI) {
#pragma unroll
        for (int kb = 0; kb < HD; kb += 32) {
            int n = lane & 15;
            int j = tI * 16 + n;
            int s = l0 - 31 + j;
            v16bf bk;
            if (s >= 0 && s < LL) {
                const __bf16* kr = Kg + (size_t)s * HD + kb + ((lane >> 4) ? 16 : 0);
                bk = cat16(*(const v8bf*)kr, *(const v8bf*)(kr + 8));
            } else {
                v8bf z = zero8b();
                bk = cat16(z, z);
            }
            S[tI] = __builtin_amdgcn_wmma_f32_16x16x32_bf16(
                false, (kb ? aq1 : aq0), false, bk, (short)0, S[tI], false, false);
        }
    }
    __syncthreads();  // VT fully staged

    // softmax over window + conn weighting + renorm; write P[m][j]
    int half = lane >> 4, nn = lane & 15;
    const float scale = 0.125f;  // 1/sqrt(64)
#pragma unroll
    for (int v = 0; v < 8; ++v) {
        int m = v + 8 * half;
        float x[3];
        bool  va[3];
#pragma unroll
        for (int tI = 0; tI < 3; ++tI) {
            int j = tI * 16 + nn;
            va[tI] = (j >= m) && (j <= m + 31);
            x[tI]  = S[tI][v] * scale;
        }
        float mx = -3.0e38f;
#pragma unroll
        for (int tI = 0; tI < 3; ++tI) if (va[tI]) mx = fmaxf(mx, x[tI]);
#pragma unroll
        for (int off = 1; off < 16; off <<= 1) mx = fmaxf(mx, __shfl_xor(mx, off, 32));
        float e[3], zs = 0.f;
#pragma unroll
        for (int tI = 0; tI < 3; ++tI) { e[tI] = va[tI] ? __expf(x[tI] - mx) : 0.f; zs += e[tI]; }
#pragma unroll
        for (int off = 1; off < 16; off <<= 1) zs += __shfl_xor(zs, off, 32);
        float f[3], fs = 0.f;
#pragma unroll
        for (int tI = 0; tI < 3; ++tI) {
            int w = tI * 16 + nn - m;
            f[tI] = va[tI] ? (e[tI] / zs) * connS[w & 31] : 0.f;
            fs += f[tI];
        }
#pragma unroll
        for (int off = 1; off < 16; off <<= 1) fs += __shfl_xor(fs, off, 32);
        float inv = 1.0f / (fs + 1e-9f);
#pragma unroll
        for (int tI = 0; tI < 3; ++tI)
            if (va[tI]) P[m][tI * 16 + nn] = (__bf16)(f[tI] * inv);
    }
    __syncthreads();

    // out = P(16x64, j-padded) x Vseg(64x48 zero-padded) via VT
    v16bf ap0 = load_a_frag(&P[0][0], 64, 0, lane);
    v16bf ap1 = load_a_frag(&P[0][0], 64, 32, lane);
    v8f O[4];
#pragma unroll
    for (int n4 = 0; n4 < 4; ++n4) O[n4] = zero8f();
#pragma unroll
    for (int n4 = 0; n4 < 4; ++n4) {
#pragma unroll
        for (int kb = 0; kb < 64; kb += 32) {
            v16bf bv = load_bt_frag(&VT[n4 * 16][0], 64, kb, lane);
            O[n4] = __builtin_amdgcn_wmma_f32_16x16x32_bf16(
                false, (kb ? ap1 : ap0), false, bv, (short)0, O[n4], false, false);
        }
    }
    // write to [B,L,D] bf16 for the output projection GEMM
#pragma unroll
    for (int n4 = 0; n4 < 4; ++n4)
#pragma unroll
        for (int v = 0; v < 8; ++v) {
            int m = v + 8 * half;
            size_t row = (size_t)b * LL + l0 + m;
            size_t col = (size_t)h * HD + n4 * 16 + nn;
            Oacc[row * DD + col] = (__bf16)O[n4][v];
        }
}

// ---------------------------------------------------------------------------
extern "C" void kernel_launch(void* const* d_in, const int* in_sizes, int n_in,
                              void* d_out, int out_size, void* d_ws, size_t ws_size,
                              hipStream_t stream) {
    const float* x    = (const float*)d_in[0];
    const float* wq   = (const float*)d_in[1];
    const float* wk   = (const float*)d_in[2];
    const float* wv   = (const float*)d_in[3];
    const float* wo   = (const float*)d_in[4];
    const float* cw1  = (const float*)d_in[5];
    const float* cw2  = (const float*)d_in[6];
    const float* cw3  = (const float*)d_in[7];
    const float* cosT = (const float*)d_in[8];
    const float* sinT = (const float*)d_in[9];

    char* ws = (char*)d_ws;
    size_t off = 0;
    auto carve = [&](size_t bytes) -> char* {
        char* p = ws + off;
        off += (bytes + 255) & ~(size_t)255;
        return p;
    };
    const size_t NTOK = (size_t)BB * LL;  // 4096
    __bf16* xb   = (__bf16*)carve(NTOK * DD * 2);
    __bf16* qraw = (__bf16*)carve(NTOK * DD * 2);
    __bf16* kraw = (__bf16*)carve(NTOK * DD * 2);
    __bf16* vraw = (__bf16*)carve(NTOK * DD * 2);
    __bf16* qr   = (__bf16*)carve(NTOK * DD * 2);
    __bf16* kr   = (__bf16*)carve(NTOK * DD * 2);
    __bf16* vr   = (__bf16*)carve(NTOK * DD * 2);
    __bf16* wqb  = (__bf16*)carve((size_t)DD * DD * 2);
    __bf16* wkb  = (__bf16*)carve((size_t)DD * DD * 2);
    __bf16* wvb  = (__bf16*)carve((size_t)DD * DD * 2);
    __bf16* wob  = (__bf16*)carve((size_t)DD * DD * 2);
    float*  connw = (float*)carve(HH * WW * 4);
    __bf16* aout = qraw;  // qraw is dead after rope_relayout; reuse for attn out

    int nx = (int)(NTOK * DD);
    f2bf_kernel<<<(nx + 255) / 256, 256, 0, stream>>>(x, xb, nx);
    int nw = DD * DD;
    f2bf_kernel<<<(nw + 255) / 256, 256, 0, stream>>>(wq, wqb, nw);
    f2bf_kernel<<<(nw + 255) / 256, 256, 0, stream>>>(wk, wkb, nw);
    f2bf_kernel<<<(nw + 255) / 256, 256, 0, stream>>>(wv, wvb, nw);
    f2bf_kernel<<<(nw + 255) / 256, 256, 0, stream>>>(wo, wob, nw);

    conn_kernel<<<1, 512, 0, stream>>>(cw1, cw2, cw3, connw);

    dim3 g(DD / 128, NTOK / 128);  // (8, 32)
    gemm_xwT<false><<<g, 256, 0, stream>>>(xb, wqb, qraw, (int)NTOK, DD, DD);
    gemm_xwT<false><<<g, 256, 0, stream>>>(xb, wkb, kraw, (int)NTOK, DD, DD);
    gemm_xwT<false><<<g, 256, 0, stream>>>(xb, wvb, vraw, (int)NTOK, DD, DD);

    int nrope = BB * LL * HH * (HD / 2);  // 2^21
    rope_relayout<<<nrope / 256, 256, 0, stream>>>(qraw, kraw, vraw, cosT, sinT, qr, kr, vr);

    attn_kernel<<<BB * HH * (LL / 16), 32, 0, stream>>>(qr, kr, vr, connw, aout);

    gemm_xwT<true><<<g, 256, 0, stream>>>(aout, wob, d_out, (int)NTOK, DD, DD);
}
